// Model_6786048327876
// MI455X (gfx1250) — matrix operand order repaired and checked by reading, not compiled, not run on hardware
//
#include <hip/hip_runtime.h>

// ---------------- problem constants ----------------
#define B_   16
#define S_   512
#define D_   768
#define V_   30522
#define H_   12
#define L_   12
#define NM_  77
#define HD_  64
#define FF_  3072
#define MROWS_ (B_ * S_)          // 8192 token rows
#define GROWS_ (B_ * NM_)         // 1232 gathered rows

// ---------------- WMMA fragment types ----------------
typedef __attribute__((ext_vector_type(16))) __bf16 bfrag16;  // A (16x32) or B (32x16) bf16 fragment
typedef __attribute__((ext_vector_type(8)))  __bf16 bvec8;
typedef __attribute__((ext_vector_type(8)))  float  facc8;    // 16x16 f32 C/D fragment

union FragU {
  bfrag16 v;
  bvec8   h[2];
  __bf16  e[16];
};

__device__ __forceinline__ facc8 wmma_bf16(bfrag16 a, bfrag16 b, facc8 c) {
  // D = A(16x32) * B(32x16) + C ; f32 accumulate
  return __builtin_amdgcn_wmma_f32_16x16x32_bf16(false, a, false, b, (short)0, c, false, false);
}

// A fragment: 16x32 (MxK). Lane L (&15) holds row M=L.
// Halves i -> K = kb + i (kb = k0 + (lane>>4)*16, 16 contiguous)
__device__ __forceinline__ bfrag16 load_a_frag(const __bf16* __restrict__ src,
                                               int row0, int k0, int lda, int maxM) {
  int lane = threadIdx.x & 31;
  int m = row0 + (lane & 15);
  if (m >= maxM) m = maxM - 1;
  int kb = k0 + ((lane >> 4) << 4);
  const __bf16* p = src + (size_t)m * lda + kb;
  FragU f;
  f.h[0] = *(const bvec8*)(p);
  f.h[1] = *(const bvec8*)(p + 8);
  return f.v;
}

// B fragment from weight-style storage W[N,K] (row n contiguous in K).
// Lane L (&15) holds column N=L; halves i -> K = k0 + (lane>>4)*16 + i (16 contiguous).
__device__ __forceinline__ bfrag16 load_bw_frag(const __bf16* __restrict__ src,
                                                int col0, int k0, int ldb, int maxN) {
  int lane = threadIdx.x & 31;
  int n = col0 + (lane & 15);
  if (n >= maxN) n = maxN - 1;
  int kb = k0 + ((lane >> 4) << 4);
  const __bf16* p = src + (size_t)n * ldb + kb;
  FragU f;
  f.h[0] = *(const bvec8*)(p);
  f.h[1] = *(const bvec8*)(p + 8);
  return f.v;
}

// Prefetch the cacheline a future B fragment will need (global_prefetch_b8).
__device__ __forceinline__ void prefetch_bw(const __bf16* __restrict__ src,
                                            int col0, int k0, int ldb, int maxN) {
  int lane = threadIdx.x & 31;
  int n = col0 + (lane & 15);
  if (n >= maxN) n = maxN - 1;
  int kb = k0 + ((lane >> 4) << 4);
  __builtin_prefetch(src + (size_t)n * ldb + kb, 0, 3);
}

// B fragment from [K,N] storage (row k contiguous in N, stride ldn) -> strided gather.
__device__ __forceinline__ bfrag16 load_bk_frag(const __bf16* __restrict__ src,
                                                int col0, int k0, int ldn) {
  int lane = threadIdx.x & 31;
  int n = col0 + (lane & 15);
  int kb = k0 + ((lane >> 4) << 4);
  FragU f;
#pragma unroll
  for (int i = 0; i < 16; ++i) f.e[i] = src[(size_t)(kb + i) * ldn + n];
  return f.v;
}

__device__ __forceinline__ float gelu_exact(float v) {
  return 0.5f * v * (1.0f + erff(v * 0.70710678118654752f));
}

// ---------------- generic WMMA GEMM:  out = A[M,K] @ W[N,K]^T + bias ----------------
// 8 waves as 4(M) x 2(N); wave tile 32x64 -> block tile 128x128.
// Lean register footprint (64 acc VGPRs) -> compiler software-pipelines loads vs WMMA.
// EPI: 0 = f32 out, 1 = bf16 out, 2 = gelu->bf16 out
template <int EPI>
__global__ void __launch_bounds__(256)
wmma_gemm_nt(const __bf16* __restrict__ A, const __bf16* __restrict__ W,
             const float* __restrict__ bias,
             float* __restrict__ outF, __bf16* __restrict__ outB,
             int M, int N, int K) {
  const int lane = threadIdx.x & 31;
  const int wv = threadIdx.x >> 5;          // 8 waves
  const int wr = wv & 3;                    // 4 wave-rows  (M)
  const int wc = wv >> 2;                   // 2 wave-cols  (N)
  const int row0 = blockIdx.x * 128 + wr * 32;
  const int col0 = blockIdx.y * 128 + wc * 64;

  facc8 acc[2][4];
#pragma unroll
  for (int i = 0; i < 2; ++i)
#pragma unroll
    for (int j = 0; j < 4; ++j)
#pragma unroll
      for (int r = 0; r < 8; ++r) acc[i][j][r] = 0.0f;

  for (int k0 = 0; k0 < K; k0 += 32) {
    bfrag16 a0 = load_a_frag(A, row0,      k0, K, M);
    bfrag16 a1 = load_a_frag(A, row0 + 16, k0, K, M);
#pragma unroll
    for (int j = 0; j < 4; ++j) {
      bfrag16 bj = load_bw_frag(W, col0 + j * 16, k0, K, N);
      if (k0 + 32 < K) prefetch_bw(W, col0 + j * 16, k0 + 32, K, N);
      acc[0][j] = wmma_bf16(a0, bj, acc[0][j]);
      acc[1][j] = wmma_bf16(a1, bj, acc[1][j]);
    }
  }

  // epilogue: C layout lane(&15)=N, (lane>>4)*8+r = M within tile
  const int n = lane & 15;
  const int mb = (lane >> 4) * 8;
#pragma unroll
  for (int i = 0; i < 2; ++i) {
#pragma unroll
    for (int j = 0; j < 4; ++j) {
      int c = col0 + j * 16 + n;
      if (c < N) {
        float bv = bias[c];
#pragma unroll
        for (int r = 0; r < 8; ++r) {
          int m = row0 + i * 16 + mb + r;
          if (m < M) {
            float v = acc[i][j][r] + bv;
            if (EPI == 0) outF[(size_t)m * N + c] = v;
            else if (EPI == 1) outB[(size_t)m * N + c] = (__bf16)v;
            else outB[(size_t)m * N + c] = (__bf16)gelu_exact(v);
          }
        }
      }
    }
  }
}

// ---------------- fused attention: one (b,h,16-row q-tile) per block ----------------
__global__ void __launch_bounds__(256)
attn_kernel(const __bf16* __restrict__ Q, const __bf16* __restrict__ Kp,
            const __bf16* __restrict__ Vp, __bf16* __restrict__ O) {
  __shared__ __attribute__((aligned(16))) float  sc[16 * 512];   // 32 KB scores
  __shared__ __attribute__((aligned(16))) __bf16 pr[16 * 512];   // 16 KB probs

  const int qt = blockIdx.x;   // 0..31
  const int h  = blockIdx.y;   // 0..11
  const int b  = blockIdx.z;   // 0..15
  const int lane = threadIdx.x & 31;
  const int wv = threadIdx.x >> 5;

  const size_t qbase = ((size_t)b * S_ + (size_t)qt * 16) * D_ + (size_t)h * HD_;
  const size_t kbase = ((size_t)b * S_) * D_ + (size_t)h * HD_;

  // ---- phase 1: scores = (Q . K^T) / 8 ; wave w covers key cols [64w, 64w+64)
  {
    facc8 acc[4];
#pragma unroll
    for (int j = 0; j < 4; ++j)
#pragma unroll
      for (int r = 0; r < 8; ++r) acc[j][r] = 0.0f;

#pragma unroll
    for (int k0 = 0; k0 < HD_; k0 += 32) {
      bfrag16 a = load_a_frag(Q + qbase, 0, k0, D_, 16);
#pragma unroll
      for (int j = 0; j < 4; ++j) {
        bfrag16 bj = load_bw_frag(Kp + kbase, wv * 64 + j * 16, k0, D_, S_);
        acc[j] = wmma_bf16(a, bj, acc[j]);
      }
    }
    const int n = lane & 15;
    const int mb = (lane >> 4) * 8;
#pragma unroll
    for (int j = 0; j < 4; ++j)
#pragma unroll
      for (int r = 0; r < 8; ++r)
        sc[(mb + r) * 512 + wv * 64 + j * 16 + n] = acc[j][r] * 0.125f;
  }
  __syncthreads();

  // ---- phase 2: softmax over 512 keys; wave w handles rows 2w, 2w+1
#pragma unroll
  for (int rr = 0; rr < 2; ++rr) {
    const int row = wv * 2 + rr;
    float vals[16];
    float mx = -3.0e38f;
#pragma unroll
    for (int i = 0; i < 16; ++i) {
      vals[i] = sc[row * 512 + lane + i * 32];
      mx = fmaxf(mx, vals[i]);
    }
    for (int o = 16; o > 0; o >>= 1) mx = fmaxf(mx, __shfl_xor(mx, o));
    float sum = 0.0f;
#pragma unroll
    for (int i = 0; i < 16; ++i) { vals[i] = __expf(vals[i] - mx); sum += vals[i]; }
    for (int o = 16; o > 0; o >>= 1) sum += __shfl_xor(sum, o);
    const float inv = 1.0f / sum;
#pragma unroll
    for (int i = 0; i < 16; ++i)
      pr[row * 512 + lane + i * 32] = (__bf16)(vals[i] * inv);
  }
  __syncthreads();

  // ---- phase 3: context = P . V ; waves 0..3 each produce a 16x16 tile of HD
  if (wv < 4) {
    facc8 acc;
#pragma unroll
    for (int r = 0; r < 8; ++r) acc[r] = 0.0f;
    for (int k0 = 0; k0 < S_; k0 += 32) {
      bfrag16 a = load_a_frag((const __bf16*)pr, 0, k0, 512, 16);
      bfrag16 bv = load_bk_frag(Vp + kbase, wv * 16, k0, D_);
      acc = wmma_bf16(a, bv, acc);
    }
    const int n = lane & 15;
    const int mb = (lane >> 4) * 8;
#pragma unroll
    for (int r = 0; r < 8; ++r) {
      size_t row = (size_t)b * S_ + (size_t)qt * 16 + mb + r;
      O[row * D_ + h * HD_ + wv * 16 + n] = (__bf16)acc[r];
    }
  }
}

// ---------------- LayerNorm(res + y) -> x (f32) and xb (bf16); wave per token ----------
__global__ void __launch_bounds__(256)
ln_kernel(const float* __restrict__ res, const float* __restrict__ y,
          const float* __restrict__ g, const float* __restrict__ be,
          float* __restrict__ xo, __bf16* __restrict__ xb) {
  const int lane = threadIdx.x & 31;
  const int wv = threadIdx.x >> 5;
  const size_t t = (size_t)blockIdx.x * 8 + wv;   // token index, grid = MROWS_/8
  const float* pr_ = res + t * D_;
  const float* py  = y + t * D_;
  float v[24];
  float sum = 0.0f;
#pragma unroll
  for (int i = 0; i < 24; ++i) {
    v[i] = pr_[lane + i * 32] + py[lane + i * 32];
    sum += v[i];
  }
  for (int o = 16; o > 0; o >>= 1) sum += __shfl_xor(sum, o);
  const float mean = sum * (1.0f / D_);
  float s2 = 0.0f;
#pragma unroll
  for (int i = 0; i < 24; ++i) { float d = v[i] - mean; s2 += d * d; }
  for (int o = 16; o > 0; o >>= 1) s2 += __shfl_xor(s2, o);
  const float inv = rsqrtf(s2 * (1.0f / D_) + 1e-5f);
#pragma unroll
  for (int i = 0; i < 24; ++i) {
    int c = lane + i * 32;
    float o = (v[i] - mean) * inv * g[c] + be[c];
    xo[t * D_ + c] = o;
    xb[t * D_ + c] = (__bf16)o;
  }
}

// ---------------- embedding sum ----------------
__global__ void __launch_bounds__(256)
embed_kernel(const int* __restrict__ tok, const int* __restrict__ seg,
             const float* __restrict__ te, const float* __restrict__ pe,
             const float* __restrict__ se, float* __restrict__ x,
             __bf16* __restrict__ xb) {
  size_t idx = (size_t)blockIdx.x * 256 + threadIdx.x;
  if (idx >= (size_t)MROWS_ * D_) return;
  size_t t = idx / D_;
  int d = (int)(idx % D_);
  int s = (int)(t % S_);
  float v = te[(size_t)tok[t] * D_ + d] + se[(size_t)seg[t] * D_ + d] + pe[(size_t)s * D_ + d];
  x[idx] = v;
  xb[idx] = (__bf16)v;
}

// ---------------- f32 -> bf16 convert ----------------
__global__ void __launch_bounds__(256)
cvt_kernel(const float* __restrict__ in, __bf16* __restrict__ out, long n) {
  long i = (long)blockIdx.x * 256 + threadIdx.x;
  if (i < n) out[i] = (__bf16)in[i];
}

// ---------------- gather masked rows (bf16) ----------------
__global__ void __launch_bounds__(256)
gather_kernel(const __bf16* __restrict__ xb, const int* __restrict__ mids,
              __bf16* __restrict__ gb) {
  size_t idx = (size_t)blockIdx.x * 256 + threadIdx.x;
  if (idx >= (size_t)GROWS_ * D_) return;
  size_t bm = idx / D_;
  int d = (int)(idx % D_);
  int b = (int)(bm / NM_);
  int s = mids[bm];
  gb[idx] = xb[(((size_t)b * S_) + s) * D_ + d];
}

// ---------------- host side ----------------
static inline void launch_gemm(int epi, const __bf16* A, const __bf16* W, const float* bias,
                               void* out, int M, int N, int K, hipStream_t s) {
  dim3 g((M + 127) / 128, (N + 127) / 128);
  dim3 blk(256);
  if (epi == 0)
    wmma_gemm_nt<0><<<g, blk, 0, s>>>(A, W, bias, (float*)out, (__bf16*)nullptr, M, N, K);
  else if (epi == 1)
    wmma_gemm_nt<1><<<g, blk, 0, s>>>(A, W, bias, (float*)nullptr, (__bf16*)out, M, N, K);
  else
    wmma_gemm_nt<2><<<g, blk, 0, s>>>(A, W, bias, (float*)nullptr, (__bf16*)out, M, N, K);
}

static inline void launch_cvt(const float* in, __bf16* out, long n, hipStream_t s) {
  long blocks = (n + 255) / 256;
  cvt_kernel<<<dim3((unsigned)blocks), dim3(256), 0, s>>>(in, out, n);
}

extern "C" void kernel_launch(void* const* d_in, const int* in_sizes, int n_in,
                              void* d_out, int out_size, void* d_ws, size_t ws_size,
                              hipStream_t stream) {
  (void)in_sizes; (void)n_in; (void)out_size; (void)ws_size;

  const int*   tok      = (const int*)d_in[0];
  const int*   seg      = (const int*)d_in[1];
  const int*   mids     = (const int*)d_in[2];
  const float* tok_emb  = (const float*)d_in[3];
  const float* pos_emb  = (const float*)d_in[4];
  const float* seg_emb  = (const float*)d_in[5];
  const float* dense_w  = (const float*)d_in[6];
  const float* dense_b  = (const float*)d_in[7];
  const float* Wq = (const float*)d_in[8];  const float* bq = (const float*)d_in[9];
  const float* Wk = (const float*)d_in[10]; const float* bk = (const float*)d_in[11];
  const float* Wv = (const float*)d_in[12]; const float* bv = (const float*)d_in[13];
  const float* Wo = (const float*)d_in[14]; const float* bo = (const float*)d_in[15];
  const float* W1 = (const float*)d_in[16]; const float* b1 = (const float*)d_in[17];
  const float* W2 = (const float*)d_in[18]; const float* b2 = (const float*)d_in[19];
  const float* g1 = (const float*)d_in[20]; const float* be1 = (const float*)d_in[21];
  const float* g2 = (const float*)d_in[22]; const float* be2 = (const float*)d_in[23];

  // ---- workspace carve (bytes, 256-aligned) ----
  char* ws = (char*)d_ws;
  size_t off = 0;
  auto carve = [&](size_t bytes) -> void* {
    void* p = ws + off;
    off += (bytes + 255) & ~(size_t)255;
    return p;
  };
  __bf16* wqb = (__bf16*)carve((size_t)D_ * D_ * 2);        // per-layer reused
  __bf16* wkb = (__bf16*)carve((size_t)D_ * D_ * 2);
  __bf16* wvb = (__bf16*)carve((size_t)D_ * D_ * 2);
  __bf16* wob = (__bf16*)carve((size_t)D_ * D_ * 2);
  __bf16* w1b = (__bf16*)carve((size_t)FF_ * D_ * 2);
  __bf16* w2b = (__bf16*)carve((size_t)FF_ * D_ * 2);
  __bf16* dwb = (__bf16*)carve((size_t)V_ * D_ * 2);
  float*  x   = (float*)carve((size_t)MROWS_ * D_ * 4);     // f32 residual stream
  __bf16* xb  = (__bf16*)carve((size_t)MROWS_ * D_ * 2);    // bf16 GEMM input
  __bf16* qb  = (__bf16*)carve((size_t)MROWS_ * D_ * 2);
  __bf16* kb_ = (__bf16*)carve((size_t)MROWS_ * D_ * 2);
  __bf16* vb_ = (__bf16*)carve((size_t)MROWS_ * D_ * 2);
  __bf16* ab  = (__bf16*)carve((size_t)MROWS_ * D_ * 2);    // attention context
  __bf16* hb  = (__bf16*)carve((size_t)MROWS_ * FF_ * 2);   // FFN hidden
  float*  yb  = (float*)carve((size_t)MROWS_ * D_ * 4);     // f32 GEMM out (pre-LN)
  __bf16* gb  = (__bf16*)carve((size_t)GROWS_ * D_ * 2);    // gathered rows

  // ---- one-time converts ----
  launch_cvt(dense_w, dwb, (long)V_ * D_, stream);

  // ---- embeddings ----
  {
    long n = (long)MROWS_ * D_;
    embed_kernel<<<dim3((unsigned)((n + 255) / 256)), dim3(256), 0, stream>>>(
        tok, seg, tok_emb, pos_emb, seg_emb, x, xb);
  }

  // ---- encoder stack ----
  for (int l = 0; l < L_; ++l) {
    launch_cvt(Wq + (size_t)l * D_ * D_, wqb, (long)D_ * D_, stream);
    launch_cvt(Wk + (size_t)l * D_ * D_, wkb, (long)D_ * D_, stream);
    launch_cvt(Wv + (size_t)l * D_ * D_, wvb, (long)D_ * D_, stream);
    launch_cvt(Wo + (size_t)l * D_ * D_, wob, (long)D_ * D_, stream);
    launch_cvt(W1 + (size_t)l * FF_ * D_, w1b, (long)FF_ * D_, stream);
    launch_cvt(W2 + (size_t)l * D_ * FF_, w2b, (long)FF_ * D_, stream);

    const float* bq_l = bq + (size_t)l * D_;
    const float* bk_l = bk + (size_t)l * D_;
    const float* bv_l = bv + (size_t)l * D_;
    const float* bo_l = bo + (size_t)l * D_;
    const float* b1_l = b1 + (size_t)l * FF_;
    const float* b2_l = b2 + (size_t)l * D_;

    // Q, K, V projections (bf16 out, [B,S,H,HD]=[B,S,D] layout)
    launch_gemm(1, xb, wqb, bq_l, qb,  MROWS_, D_, D_, stream);
    launch_gemm(1, xb, wkb, bk_l, kb_, MROWS_, D_, D_, stream);
    launch_gemm(1, xb, wvb, bv_l, vb_, MROWS_, D_, D_, stream);

    // fused attention
    attn_kernel<<<dim3(S_ / 16, H_, B_), dim3(256), 0, stream>>>(qb, kb_, vb_, ab);

    // output projection (f32 out), then LN1 (residual add inside)
    launch_gemm(0, ab, wob, bo_l, yb, MROWS_, D_, D_, stream);
    ln_kernel<<<dim3(MROWS_ / 8), dim3(256), 0, stream>>>(x, yb, g1 + (size_t)l * D_,
                                                          be1 + (size_t)l * D_, x, xb);

    // FFN: GELU(x W1^T + b1) W2^T + b2, then LN2
    launch_gemm(2, xb, w1b, b1_l, hb, MROWS_, FF_, D_, stream);
    launch_gemm(0, hb, w2b, b2_l, yb, MROWS_, D_, FF_, stream);
    ln_kernel<<<dim3(MROWS_ / 8), dim3(256), 0, stream>>>(x, yb, g2 + (size_t)l * D_,
                                                          be2 + (size_t)l * D_, x, xb);
  }

  // ---- gather masked positions, MLM head ----
  {
    long n = (long)GROWS_ * D_;
    gather_kernel<<<dim3((unsigned)((n + 255) / 256)), dim3(256), 0, stream>>>(xb, mids, gb);
  }
  launch_gemm(0, gb, dwb, dense_b, d_out, GROWS_, V_, D_, stream);
}
